// infonce_loss_73220602462632
// MI455X (gfx1250) — matrix operand order repaired and checked by reading, not compiled, not run on hardware
//
#include <hip/hip_runtime.h>

typedef __attribute__((ext_vector_type(16))) _Float16 v16h;
typedef __attribute__((ext_vector_type(8)))  _Float16 v8h;
typedef __attribute__((ext_vector_type(8)))  float    v8f;

#define N_ROWS   4096
#define ZDIM     256
#define TWO_N    8192
#define INV_T    (1.0f / 0.07f)
// exp(INV_T*(x-1)) == exp2(C2*x - C2), C2 = log2(e)/T
// x in [-1,1] -> argument in [-41.3, 0]: raw v_exp_f32 is safe (no denorm fixup)
#define C2       (1.4426950408889634f * INV_T)
#define NTILES   (TWO_N / 16)      // 512 column tiles
#define HALF_T   (NTILES / 2)      // 256 tiles per column-half wave

// ---------------------------------------------------------------------------
// Kernel 1: L2-normalize rows of z1 (rows 0..4095) and z2 (rows 4096..8191)
// into a row-major f16 matrix reps[8192][256]. One wave per row.
// ---------------------------------------------------------------------------
__global__ void nrm_kernel(const float* __restrict__ z1,
                           const float* __restrict__ z2,
                           _Float16* __restrict__ reps) {
  const int wave = threadIdx.x >> 5;
  const int lane = threadIdx.x & 31;
  const int row  = blockIdx.x * 8 + wave;

  const float* __restrict__ src =
      (row < N_ROWS) ? (z1 + (size_t)row * ZDIM)
                     : (z2 + (size_t)(row - N_ROWS) * ZDIM);

  float4 f0 = ((const float4*)src)[lane * 2 + 0];
  float4 f1 = ((const float4*)src)[lane * 2 + 1];

  float ss = f0.x * f0.x + f0.y * f0.y + f0.z * f0.z + f0.w * f0.w +
             f1.x * f1.x + f1.y * f1.y + f1.z * f1.z + f1.w * f1.w;
#pragma unroll
  for (int m = 16; m >= 1; m >>= 1) ss += __shfl_xor(ss, m, 32);

  const float inv = 1.0f / fmaxf(sqrtf(ss), 1e-12f);

  v8h h;
  h[0] = (_Float16)(f0.x * inv); h[1] = (_Float16)(f0.y * inv);
  h[2] = (_Float16)(f0.z * inv); h[3] = (_Float16)(f0.w * inv);
  h[4] = (_Float16)(f1.x * inv); h[5] = (_Float16)(f1.y * inv);
  h[6] = (_Float16)(f1.z * inv); h[7] = (_Float16)(f1.w * inv);
  *(v8h*)(reps + (size_t)row * ZDIM + lane * 8) = h;   // 16B store
}

// ---------------------------------------------------------------------------
__device__ __forceinline__ v16h cat8(v8h lo, v8h up) {
  return __builtin_shufflevector(lo, up, 0, 1, 2, 3, 4, 5, 6, 7,
                                         8, 9, 10, 11, 12, 13, 14, 15);
}

// raw hardware exp2 (v_exp_f32), argument range is statically safe here
__device__ __forceinline__ float fast_exp2(float x) {
  return __builtin_amdgcn_exp2f(x);
}

// B fragments for one 16-col tile: lane holds column (lane16), 16 contiguous
// K halves starting at kb + hi*16 (contiguous K order, same as A).
__device__ __forceinline__ void load_btile(v16h b[8], const _Float16* p) {
#pragma unroll
  for (int k = 0; k < 8; ++k) {
    v8h lo = *(const v8h*)(p + 32 * k);       // global_load_b128
    v8h up = *(const v8h*)(p + 32 * k + 8);   // global_load_b128
    b[k] = cat8(lo, up);
  }
}

// 8 chained WMMAs for one 16x16 sim tile + fused fixed-shift LSE epilogue.
// Self-diag / positive handling only in their (scalar-uniform) special tiles.
__device__ __forceinline__ void tile_compute(const v16h a[8], const v16h b[8],
                                             int ct, int ctSelf, int ctPos,
                                             float s[8], float& posv,
                                             int lane16, int hi) {
  v8f c = {};
#pragma unroll
  for (int k = 0; k < 8; ++k)
    c = __builtin_amdgcn_wmma_f32_16x16x32_f16(
            false, a[k], false, b[k], (short)0, c, false, false);

  // hot epilogue: exp2(C2*x - C2) <= 1 always (cosine bound) -> exactly stable
  // one v_pk_fma_f32 per pair + one v_exp_f32 per element
#pragma unroll
  for (int r = 0; r < 8; ++r)
    s[r] += fast_exp2(fmaf(c[r], C2, -C2));

  if (ct == ctSelf) {            // scalar-uniform: tile containing the diagonal
#pragma unroll
    for (int r = 0; r < 8; ++r)
      if (lane16 == r + 8 * hi)  // lane owning sim[row,row]
        s[r] -= fast_exp2(fmaf(c[r], C2, -C2));
  }
  if (ct == ctPos) {             // scalar-uniform: tile containing positives
#pragma unroll
    for (int r = 0; r < 8; ++r)
      if (lane16 == r + 8 * hi)  // partner col = row ^ 4096 -> same low 4 bits
        posv += c[r];
  }
}

// ---------------------------------------------------------------------------
// Kernel 2: fused sim = reps @ reps^T + row-wise logsumexp + positive gather.
// Block = 256 threads = 8 waves = 4 row-strips x 2 column-halves.
// Each wave: 16 rows register-resident (A, 64 VGPRs), sweeps 256 column tiles
// with register double-buffered B (prefetch tile ct+1 while computing ct).
// ---------------------------------------------------------------------------
__global__ void __launch_bounds__(256, 1)
simclr_kernel(const _Float16* __restrict__ reps, float* __restrict__ partials) {
  const int lane   = threadIdx.x & 31;
  const int lane16 = lane & 15;
  const int hi     = lane >> 4;                  // half-wave (0/1)
  const int wv     = threadIdx.x >> 5;           // wave in block (0..7)
  // force wave-uniform control values into SGPRs -> scalar branches in loop
  const int strip  = __builtin_amdgcn_readfirstlane(blockIdx.x * 4 + (wv >> 1));
  const int half   = __builtin_amdgcn_readfirstlane(wv & 1);
  const int rowBase = strip << 4;
  const int ctSelf  = strip;                     // tile holding the diagonal
  const int ctPos   = strip ^ (N_ROWS >> 4);     // tile holding positives

  // ---- A fragments (contiguous K: lanes 0-15 K{0..15}, lanes 16-31 K{16..31})
  v16h a[8];
  {
    const _Float16* ap = reps + (size_t)(rowBase + lane16) * ZDIM + hi * 16;
#pragma unroll
    for (int k = 0; k < 8; ++k)
      a[k] = cat8(*(const v8h*)(ap + 32 * k),
                  *(const v8h*)(ap + 32 * k + 8));
  }

  float s[8];
#pragma unroll
  for (int r = 0; r < 8; ++r) s[r] = 0.0f;
  float posv = 0.0f;

  const int ctBeg = half * HALF_T;
  const int ctEnd = ctBeg + HALF_T;
  const _Float16* bbase = reps + (size_t)lane16 * ZDIM + hi * 16;

  // ---- register double-buffered column-tile sweep (unroll by 2)
  v16h b0[8], b1[8];
  load_btile(b0, bbase + (size_t)ctBeg * (16 * ZDIM));
  for (int ct = ctBeg; ct < ctEnd; ct += 2) {
    load_btile(b1, bbase + (size_t)(ct + 1) * (16 * ZDIM));
    tile_compute(a, b0, ct, ctSelf, ctPos, s, posv, lane16, hi);
    const int ctn = (ct + 2 < ctEnd) ? (ct + 2) : ct;   // clamp: no OOB read
    load_btile(b0, bbase + (size_t)ctn * (16 * ZDIM));
    tile_compute(a, b1, ct + 1, ctSelf, ctPos, s, posv, lane16, hi);
  }

  // ---- merge the two column-half waves of each strip through LDS
  __shared__ float red_s[8][32][8];
  __shared__ float red_p[8][32];
  if (half == 1) {
#pragma unroll
    for (int r = 0; r < 8; ++r) red_s[wv][lane][r] = s[r];
    red_p[wv][lane] = posv;
  }
  __syncthreads();
  if (half == 0) {
#pragma unroll
    for (int r = 0; r < 8; ++r) s[r] += red_s[wv + 1][lane][r];
    posv += red_p[wv + 1][lane];

    // per-row totals: sum 16 column-subsets across each 16-lane half
    float part = 0.0f;
#pragma unroll
    for (int r = 0; r < 8; ++r) {
      float sv = s[r];
#pragma unroll
      for (int m = 8; m >= 1; m >>= 1) sv += __shfl_xor(sv, m, 32);
      part += __logf(sv);
    }
    float pv = posv;
#pragma unroll
    for (int m = 8; m >= 1; m >>= 1) pv += __shfl_xor(pv, m, 32);
    // Sum over 8 rows of: (1/T + log s) - pos/T  ==  8/T - pv/T + sum(log s)
    part += INV_T * (8.0f - pv);
    if (lane16 == 0) partials[strip * 2 + hi] = part;
  }
}

// ---------------------------------------------------------------------------
// Kernel 3: deterministic tree reduction of 1024 partials -> mean loss scalar.
// ---------------------------------------------------------------------------
__global__ void fin_kernel(const float* __restrict__ partials,
                           float* __restrict__ out) {
  __shared__ float smem[32];
  const int t    = threadIdx.x;
  const int lane = t & 31;
  const int wave = t >> 5;

  float v = partials[t];                         // blockDim.x == 1024
#pragma unroll
  for (int m = 16; m >= 1; m >>= 1) v += __shfl_xor(v, m, 32);
  if (lane == 0) smem[wave] = v;
  __syncthreads();
  if (wave == 0) {
    float w = smem[lane];
#pragma unroll
    for (int m = 16; m >= 1; m >>= 1) w += __shfl_xor(w, m, 32);
    if (lane == 0) out[0] = w / (float)TWO_N;
  }
}

// ---------------------------------------------------------------------------
extern "C" void kernel_launch(void* const* d_in, const int* in_sizes, int n_in,
                              void* d_out, int out_size, void* d_ws, size_t ws_size,
                              hipStream_t stream) {
  const float* z1 = (const float*)d_in[0];
  const float* z2 = (const float*)d_in[1];
  float* out = (float*)d_out;

  // workspace layout: [ reps f16: 8192*256*2 B = 4 MiB | partials: 1024 f32 ]
  _Float16* reps = (_Float16*)d_ws;
  float* partials =
      (float*)((char*)d_ws + (size_t)TWO_N * ZDIM * sizeof(_Float16));

  nrm_kernel<<<TWO_N / 8, 256, 0, stream>>>(z1, z2, reps);
  simclr_kernel<<<128, 256, 0, stream>>>(reps, partials);
  fin_kernel<<<1, 1024, 0, stream>>>(partials, out);
}
